// SelectiveMLP_80994493268163
// MI455X (gfx1250) — matrix operand order repaired and checked by reading, not compiled, not run on hardware
//
#include <hip/hip_runtime.h>

// SelectiveMLP on gfx1250 (MI455X): bf16 WMMA, f32 accumulation, LDS-staged.
//   xb  [M,K]    = bf16(x)
//   w1b [IDX,K]  = bf16(fc1_w[idx])          (gathered, dense)
//   w2t [K,IDX]  = bf16(fc2_w[idx])^T        (gathered + transposed)
//   h   [M,IDX]  = relu(xb @ w1b^T)          (bf16)
//   out [M,K]    = h @ w2t^T                 (f32)
// GEMM: 256-thread workgroup (8 waves) owns a 128x128 tile.  Shared 128x32
// B tiles are double-buffered through LDS; global loads run one full k-step
// ahead of their ds_store.  Workgroup barriers are LDS-fence only
// (s_wait_dscnt 0 + split barrier) so in-flight global loads are NOT drained
// at the barrier -- their latency stays hidden under a whole iteration.
// Workspace: 4 x 32MB = 128MB.

#define MDIM 4096
#define KDIM 4096
#define IDXN 4096

#define LROW 40                 // padded LDS row (elements): 80B, bank-friendly
#define LBUF (128 * LROW)       // one LDS buffer (elements)

typedef __attribute__((ext_vector_type(16))) __bf16 v16bf;
typedef __attribute__((ext_vector_type(8)))  __bf16 v8bf;
typedef __attribute__((ext_vector_type(8)))  float  v8f;

// Workgroup barrier with LDS-only fence: waits for this wave's DS ops, then
// split-barrier.  Leaves global LOADcnt outstanding (unlike __syncthreads()).
__device__ inline void lds_barrier() {
    asm volatile(
        "s_wait_dscnt 0x0\n\t"
        "s_barrier_signal -1\n\t"
        "s_barrier_wait -1"
        ::: "memory");
}

__device__ inline v8bf cvt8(float4 v0, float4 v1) {
    v8bf o;
    o[0] = (__bf16)v0.x; o[1] = (__bf16)v0.y; o[2] = (__bf16)v0.z; o[3] = (__bf16)v0.w;
    o[4] = (__bf16)v1.x; o[5] = (__bf16)v1.y; o[6] = (__bf16)v1.z; o[7] = (__bf16)v1.w;
    return o;
}

// A fragment (16x32 bf16): two adjacent 16B chunks (16 contiguous k).
__device__ inline v16bf loadA(const __bf16* p) {
    const v8bf a0 = *(const v8bf*)(p);
    const v8bf a1 = *(const v8bf*)(p + 8);
    v16bf a;
#pragma unroll
    for (int i = 0; i < 8; ++i) { a[i] = a0[i]; a[i + 8] = a1[i]; }
    return a;
}

// ---------------------------------------------------------------------------
// Pre-pass A: xb = bf16(x).
// ---------------------------------------------------------------------------
__global__ __launch_bounds__(256)
void selmlp_cvt_x(const float* __restrict__ x, __bf16* __restrict__ xb) {
    const size_t i = ((size_t)blockIdx.x * 256 + threadIdx.x) * 8;
    const float4 v0 = *(const float4*)(x + i);
    const float4 v1 = *(const float4*)(x + i + 4);
    *(v8bf*)(xb + i) = cvt8(v0, v1);
}

// ---------------------------------------------------------------------------
// Pre-pass B: w1b[n][k] = bf16(fc1_w[idx[n]][k]).  Row-coalesced gather.
// ---------------------------------------------------------------------------
__global__ __launch_bounds__(256)
void selmlp_gather1(const float* __restrict__ fc1_w, const int* __restrict__ idx,
                    __bf16* __restrict__ w1b) {
    const int n = blockIdx.x;
    const int k = blockIdx.y * 2048 + threadIdx.x * 8;
    const float* src = fc1_w + (size_t)idx[n] * KDIM + k;
    const float4 v0 = *(const float4*)(src);
    const float4 v1 = *(const float4*)(src + 4);
    *(v8bf*)(w1b + (size_t)n * KDIM + k) = cvt8(v0, v1);
}

// ---------------------------------------------------------------------------
// Pre-pass C: w2t[k][n] = bf16(fc2_w[idx[n]][k]).  64x64 LDS-tiled transpose.
// ---------------------------------------------------------------------------
__global__ __launch_bounds__(256)
void selmlp_gather_t(const float* __restrict__ fc2_w, const int* __restrict__ idx,
                     __bf16* __restrict__ w2t) {
    __shared__ __bf16 tile[64][64 + 2];
    const int n0 = blockIdx.x * 64;
    const int k0 = blockIdx.y * 64;
    const int t  = threadIdx.x;
    {
        const int nl = t >> 2;
        const int kq = (t & 3) * 16;
        const float* src = fc2_w + (size_t)idx[n0 + nl] * KDIM + k0 + kq;
#pragma unroll
        for (int j = 0; j < 16; j += 4) {
            const float4 v = *(const float4*)(src + j);
            tile[nl][kq + j + 0] = (__bf16)v.x;
            tile[nl][kq + j + 1] = (__bf16)v.y;
            tile[nl][kq + j + 2] = (__bf16)v.z;
            tile[nl][kq + j + 3] = (__bf16)v.w;
        }
    }
    __syncthreads();
    {
        const int kl = t >> 2;
        const int nq = (t & 3) * 16;
        __bf16 tmp[16];
#pragma unroll
        for (int j = 0; j < 16; ++j) tmp[j] = tile[nq + j][kl];
        __bf16* dst = w2t + (size_t)(k0 + kl) * IDXN + n0 + nq;
        *(uint4*)dst = *(const uint4*)tmp;
    }
}

// ---------------------------------------------------------------------------
// GEMM1: h = relu(xb @ w1b^T), bf16 out.  128x128 tile per 256-thread block.
// ---------------------------------------------------------------------------
__global__ __launch_bounds__(256)
void selmlp_gemm1(const __bf16* __restrict__ xb, const __bf16* __restrict__ w1b,
                  __bf16* __restrict__ hbuf) {
    __shared__ __bf16 lds[2 * LBUF];
    const int t    = threadIdx.x;
    const int lane = t & 31;
    const int wv   = t >> 5;                  // wave id 0..7
    const int l16  = lane & 15;
    const int half = lane >> 4;
    const int n0   = blockIdx.x * 128;
    const int m0   = blockIdx.y * 128 + wv * 16;

    const __bf16* abase = xb + (size_t)(m0 + l16) * KDIM + half * 16;

    // B-tile staging coords: thread t moves row n_local, 16-elem k-chunk koff.
    const int n_local = t >> 1;
    const int koff    = (t & 1) * 16;
    const __bf16* gsrc = w1b + (size_t)(n0 + n_local) * KDIM + koff;

    v8f acc[8] = {};

    // Prologue: stage step 0 into buffer 0; preload step 1 into registers.
    *(v16bf*)(&lds[n_local * LROW + koff]) = *(const v16bf*)(gsrc);
    v16bf a_cur = loadA(abase);
    v16bf g_nxt = *(const v16bf*)(gsrc + 32);
    v16bf a_nxt = loadA(abase + 32);
    lds_barrier();

    for (int k0 = 0; k0 < KDIM; k0 += 32) {
        const int cur = (k0 >> 5) & 1;
        const int nxt = cur ^ 1;
        const int k2  = (k0 + 64 < KDIM) ? (k0 + 64) : 0;  // two ahead; wrap

        __builtin_prefetch(gsrc + k0 + 96, 0, 1);
        // Issue loads for step k0+64 (consumed one iteration from now).
        const v16bf g_far = *(const v16bf*)(gsrc + k2);
        const v16bf a_far = loadA(abase + k2);

        const __bf16* lbase = &lds[cur * LBUF + l16 * LROW + half * 16];
#pragma unroll
        for (int s = 0; s < 8; ++s) {
            const v16bf b = *(const v16bf*)(lbase + s * (16 * LROW));
            acc[s] = __builtin_amdgcn_wmma_f32_16x16x32_bf16(
                false, a_cur, false, b, (short)0, acc[s], false, false);
        }

        // Store step k0+32 tile (its load finished an iteration ago).
        *(v16bf*)(&lds[nxt * LBUF + n_local * LROW + koff]) = g_nxt;
        lds_barrier();

        a_cur = a_nxt;
        a_nxt = a_far;
        g_nxt = g_far;
    }

    // C/D layout: VGPR r -> M = r + half*8, N = l16 (per sub-tile).
#pragma unroll
    for (int s = 0; s < 8; ++s) {
        const int n = n0 + s * 16 + l16;
#pragma unroll
        for (int r = 0; r < 8; ++r) {
            const int m = m0 + r + half * 8;
            const float v = acc[s][r];
            hbuf[(size_t)m * IDXN + n] = (__bf16)(v > 0.0f ? v : 0.0f);
        }
    }
}

// ---------------------------------------------------------------------------
// GEMM2: out[m,k] = sum_n h[m,n] * w2t[k][n], f32 out.  Mirror of GEMM1.
// ---------------------------------------------------------------------------
__global__ __launch_bounds__(256)
void selmlp_gemm2(const __bf16* __restrict__ hbuf, const __bf16* __restrict__ w2t,
                  float* __restrict__ out) {
    __shared__ __bf16 lds[2 * LBUF];
    const int t    = threadIdx.x;
    const int lane = t & 31;
    const int wv   = t >> 5;
    const int l16  = lane & 15;
    const int half = lane >> 4;
    const int k0c  = blockIdx.x * 128;        // output-column tile base
    const int m0   = blockIdx.y * 128 + wv * 16;

    const __bf16* abase = hbuf + (size_t)(m0 + l16) * IDXN + half * 16;

    const int c_local = t >> 1;               // output column within tile
    const int noff    = (t & 1) * 16;         // reduction-chunk offset
    const __bf16* gsrc = w2t + (size_t)(k0c + c_local) * IDXN + noff;

    v8f acc[8] = {};

    *(v16bf*)(&lds[c_local * LROW + noff]) = *(const v16bf*)(gsrc);
    v16bf a_cur = loadA(abase);
    v16bf g_nxt = *(const v16bf*)(gsrc + 32);
    v16bf a_nxt = loadA(abase + 32);
    lds_barrier();

    for (int n0 = 0; n0 < IDXN; n0 += 32) {
        const int cur = (n0 >> 5) & 1;
        const int nxt = cur ^ 1;
        const int n2  = (n0 + 64 < IDXN) ? (n0 + 64) : 0;

        __builtin_prefetch(gsrc + n0 + 96, 0, 1);
        const v16bf g_far = *(const v16bf*)(gsrc + n2);
        const v16bf a_far = loadA(abase + n2);

        const __bf16* lbase = &lds[cur * LBUF + l16 * LROW + half * 16];
#pragma unroll
        for (int s = 0; s < 8; ++s) {
            const v16bf b = *(const v16bf*)(lbase + s * (16 * LROW));
            acc[s] = __builtin_amdgcn_wmma_f32_16x16x32_bf16(
                false, a_cur, false, b, (short)0, acc[s], false, false);
        }

        *(v16bf*)(&lds[nxt * LBUF + c_local * LROW + noff]) = g_nxt;
        lds_barrier();

        a_cur = a_nxt;
        a_nxt = a_far;
        g_nxt = g_far;
    }

#pragma unroll
    for (int s = 0; s < 8; ++s) {
        const int k = k0c + s * 16 + l16;
#pragma unroll
        for (int r = 0; r < 8; ++r) {
            const int m = m0 + r + half * 8;
            out[(size_t)m * KDIM + k] = acc[s][r];
        }
    }
}

extern "C" void kernel_launch(void* const* d_in, const int* in_sizes, int n_in,
                              void* d_out, int out_size, void* d_ws, size_t ws_size,
                              hipStream_t stream) {
    const float* x     = (const float*)d_in[0];
    const float* fc1_w = (const float*)d_in[1];
    const float* fc2_w = (const float*)d_in[2];
    const int*   idx   = (const int*)d_in[3];
    float*       out   = (float*)d_out;

    const size_t SEG = (size_t)MDIM * IDXN * sizeof(__bf16);   // 32 MB
    __bf16* xbuf = (__bf16*)((char*)d_ws + 0 * SEG);
    __bf16* w1b  = (__bf16*)((char*)d_ws + 1 * SEG);
    __bf16* w2t  = (__bf16*)((char*)d_ws + 2 * SEG);
    __bf16* hbuf = (__bf16*)((char*)d_ws + 3 * SEG);

    // Pre-pass (all independent; stream-serialized before the GEMMs).
    selmlp_cvt_x   <<<dim3((MDIM * (size_t)KDIM) / 2048), 256, 0, stream>>>(x, xbuf);
    selmlp_gather1 <<<dim3(IDXN, KDIM / 2048), 256, 0, stream>>>(fc1_w, idx, w1b);
    selmlp_gather_t<<<dim3(IDXN / 64, KDIM / 64), 256, 0, stream>>>(fc2_w, idx, w2t);

    // GEMM1: h = relu(xb @ w1b^T)
    selmlp_gemm1<<<dim3(IDXN / 128, MDIM / 128), 256, 0, stream>>>(xbuf, w1b, hbuf);

    // GEMM2: out = h @ w2t^T
    selmlp_gemm2<<<dim3(KDIM / 128, MDIM / 128), 256, 0, stream>>>(hbuf, w2t, out);
}
